// Quantize_13340168421720
// MI455X (gfx1250) — matrix operand order repaired and checked by reading, not compiled, not run on hardware
//
#include <hip/hip_runtime.h>

typedef __attribute__((ext_vector_type(16))) __bf16 v16bf;
typedef __attribute__((ext_vector_type(8)))  __bf16 v8bf;
typedef __attribute__((ext_vector_type(8)))  float  v8f;
typedef __attribute__((ext_vector_type(4)))  unsigned int v4u;
typedef __attribute__((ext_vector_type(8)))  int    v8i;
typedef __attribute__((ext_vector_type(4)))  int    v4i;

#define DIM        256
#define KCODES     8192
#define NROWS      16384           // 16 * 32 * 32
#define HW         1024            // 32*32
#define RPB        128             // rows per block (8 waves x 16 rows)

// combined codebook layout: per code k, 512B of hi-bf16 then 512B of lo-bf16
#define COLB       1024            // bytes per code column in global etc
#define COLB_LDS   1040            // bytes per code column in LDS (TDM pad 16B/1024B)
#define CHUNK_COLS 16
#define CHUNK_GB   (CHUNK_COLS * COLB)      // 16384 bytes per chunk in global
#define CHUNK_LDS  (CHUNK_COLS * COLB_LDS)  // 16640 bytes per chunk in LDS
#define NCHUNK     (KCODES / CHUNK_COLS)    // 512

// d_out layout (flat, f32): quantize | diff | embed_ind | perplexity
#define QUANT_OFF  0
#define DIFF_OFF   4194304
#define IND_OFF    4194305
#define PERP_OFF   4210689

#if defined(__AMDGCN__) && __has_builtin(__builtin_amdgcn_tensor_load_to_lds)
#define VQ_TDM 1
#if __has_include(<hip/amd_detail/amd_gfx1250_TDM.h>)
#define VQ_TDM_6ARG 1
#else
#define VQ_TDM_6ARG 0
#endif
#else
#define VQ_TDM 0
#endif

#if defined(__AMDGCN__) && __has_builtin(__builtin_amdgcn_sched_group_barrier)
#define VQ_SCHED 1
#else
#define VQ_SCHED 0
#endif

// ---------------------------------------------------------------- init
__global__ void vq_init(float* __restrict__ counts, float* __restrict__ sse) {
  int t = blockIdx.x * blockDim.x + threadIdx.x;
  if (t < KCODES) counts[t] = 0.0f;
  if (t == 0) *sse = 0.0f;
}

// --------------------------- transpose + bf16 hi/lo split (combined layout)
// e[C,K] f32 -> etc[K][hi bf16 x256 | lo bf16 x256]
__global__ void vq_prep_codes(const float* __restrict__ e,
                              __bf16* __restrict__ etc) {
  __shared__ float tile[32][33];
  const int k0 = blockIdx.x * 32;
  const int c0 = blockIdx.y * 32;
  const int t  = threadIdx.x;
#pragma unroll
  for (int p = 0; p < 4; ++p) {
    int idx = p * 256 + t;
    int kk = idx & 31, cc = idx >> 5;
    tile[cc][kk] = e[(size_t)(c0 + cc) * KCODES + k0 + kk];
  }
  __syncthreads();
#pragma unroll
  for (int p = 0; p < 4; ++p) {
    int idx = p * 256 + t;
    int cc = idx & 31, kk = idx >> 5;
    float v = tile[cc][kk];
    __bf16 hi = (__bf16)v;
    __bf16 lo = (__bf16)(v - (float)hi);
    size_t base = (size_t)(k0 + kk) * 512;   // 512 bf16 elements per column
    etc[base + c0 + cc]       = hi;
    etc[base + 256 + c0 + cc] = lo;
  }
}

// ---------------------------------------------------------------- ||e_k||^2
__global__ void vq_colnorm(const float* __restrict__ e, float* __restrict__ cn) {
  int k = blockIdx.x * blockDim.x + threadIdx.x;
  float s = 0.0f;
  for (int c = 0; c < DIM; ++c) {
    float v = e[(size_t)c * KCODES + k];
    s += v * v;
  }
  cn[k] = s;
}

#if VQ_TDM
// Issue one 2-D TDM copy: 16 codebook columns (16 KB) -> LDS, with the TDM
// inserting 16B of pad per 1024B so LDS column stride is 1040B (bank-friendly).
__device__ __forceinline__ void vq_tdm_issue(unsigned lds_off,
                                             unsigned long long gaddr) {
  v4u g0;
  g0[0] = 1u;                                   // count=1 (valid), user mode
  g0[1] = lds_off;                              // LDS byte address
  g0[2] = (unsigned)gaddr;                      // global addr [31:0]
  g0[3] = (unsigned)((gaddr >> 32) & 0x1FFFFFFu) | (2u << 30);  // [56:32] | type=2
  v8i g1;
  g1[0] = (int)((3u << 16)      // data_size = 8B
              | (1u << 20)      // pad_enable
              | (7u << 22)      // pad_interval: 256 dwords (1024B)
              | (3u << 25));    // pad_amount: 4 dwords (16B)
  g1[1] = (int)(128u << 16);    // tensor_dim0 = 128 (x 8B = 1024B row)
  g1[2] = (int)(16u << 16);     // tensor_dim1 = 16 rows
  g1[3] = (int)(128u << 16);    // tile_dim0 = 128
  g1[4] = 16;                   // tile_dim1 = 16
  g1[5] = 128;                  // tensor_dim0_stride = 128 elements
  g1[6] = 0;
  g1[7] = 0;
  v4i gz = {0, 0, 0, 0};
#if VQ_TDM_6ARG
  v8i gz8 = {0, 0, 0, 0, 0, 0, 0, 0};
  __builtin_amdgcn_tensor_load_to_lds(g0, g1, gz, gz, gz8, 0);
#else
  __builtin_amdgcn_tensor_load_to_lds(g0, g1, gz, gz, 0);
#endif
}
__device__ __forceinline__ void vq_wait_tensor0() {
#if __has_builtin(__builtin_amdgcn_s_wait_tensorcnt)
  __builtin_amdgcn_s_wait_tensorcnt(0);
#else
  asm volatile("s_wait_tensorcnt 0x0" ::: "memory");
#endif
}
#endif

// ---------------------------------------------------------------- main
// Block = 128 rows (wave w owns rows [w*16,w*16+16)). All 8 waves march over
// the codebook in lockstep, 16 columns per chunk, B staged in LDS by the TDM
// (double-buffered, copy of chunk i+1 overlaps WMMA on chunk i). The
// DS-read / WMMA pipeline shape is pinned with sched_group_barrier so B
// fragment loads lead their consumers by one stage (wait dscnt<=4, not 0).
__global__ __launch_bounds__(256) void vq_main(
    const float* __restrict__ x, const float* __restrict__ e,
    const __bf16* __restrict__ etc, const float* __restrict__ colnorm,
    float* __restrict__ out, float* __restrict__ counts,
    float* __restrict__ sse_acc) {
  extern __shared__ char smem[];
  char* bbuf = smem;                                         // 2 * 16640 B
  float (*xs)[260] = (float(*)[260])(smem + 2 * CHUNK_LDS);  // 16640 B
  float* red   = (float*)(smem + 2 * CHUNK_LDS + 16640);     // 1024 B
  int*   resIdx = (int*)(smem + 2 * CHUNK_LDS + 16640 + 1024); // 512 B

  const int t    = threadIdx.x;
  const int lane = t & 31;
  const int wave = t >> 5;
  const int R0   = blockIdx.x * RPB;
  const int b    = R0 >> 10;          // 128 | 1024 -> whole block same batch
  const int hw0  = R0 & 1023;

  // ---- build A fragments (bf16 hi/lo, ISA 16-bit A layout), one wave at a time
  v16bf a_hi[8], a_lo[8];
  for (int w = 0; w < 8; ++w) {
#pragma unroll
    for (int j = 0; j < 16; ++j) {
      int m = t & 15;
      int c = (t >> 4) + j * 16;
      xs[m][c] = x[((size_t)(b * DIM + c)) * HW + hw0 + w * 16 + m];
    }
    __syncthreads();
    if (wave == w) {
      const int m  = lane & 15;
      const int kb = (lane < 16) ? 0 : 8;  // A: lanes 0-15 K{0-7,16-23}, 16-31 K{8-15,24-31}
#pragma unroll
      for (int kt = 0; kt < 8; ++kt) {
        v16bf h, l;
#pragma unroll
        for (int i = 0; i < 16; ++i) {
          int kk = kt * 32 + kb + (i < 8 ? i : i + 8);
          float v   = xs[m][kk];
          __bf16 hi = (__bf16)v;
          __bf16 lo = (__bf16)(v - (float)hi);
          h[i] = hi; l[i] = lo;
        }
        a_hi[kt] = h; a_lo[kt] = l;
      }
    }
    __syncthreads();
  }

  // ---- argmin state
  float minv[8];
  int   mini[8];
#pragma unroll
  for (int v = 0; v < 8; ++v) { minv[v] = __builtin_inff(); mini[v] = 0; }

  const int nlane = lane & 15;
  const int chalf = (lane >> 4) << 3;   // B: lanes 0-15 -> c {0-7,16-23}, lanes 16-31 -> c {8-15,24-31}

#if VQ_TDM
  const unsigned lds0 = __builtin_amdgcn_groupstaticsize();
  const unsigned long long gbase = (unsigned long long)(uintptr_t)etc;
  if (wave == 0) vq_tdm_issue(lds0, gbase);                  // prologue: chunk 0
#endif

  for (int ct = 0; ct < NCHUNK; ++ct) {
    const char* bb = bbuf + (ct & 1) * CHUNK_LDS;

#if VQ_TDM
    if (wave == 0) vq_wait_tensor0();    // chunk ct landed in LDS
    __syncthreads();                     // visible to all; prev reads done
    if (wave == 0 && ct + 1 < NCHUNK)    // prefetch next chunk into other buffer
      vq_tdm_issue(lds0 + (unsigned)(((ct + 1) & 1) * CHUNK_LDS),
                   gbase + (unsigned long long)(ct + 1) * CHUNK_GB);
#else
    {
      // fallback: cooperative synchronous copy (256 threads x 16B x 4)
      const char* gs = (const char*)etc + (size_t)ct * CHUNK_GB;
      char* db = bbuf + (ct & 1) * CHUNK_LDS;
#pragma unroll
      for (int i = 0; i < 4; ++i) {
        int idx  = t * 16 + i * 4096;
        int ccol = idx >> 10, off = idx & 1023;
        *(v4u*)(db + ccol * COLB_LDS + off) = *(const v4u*)(gs + idx);
      }
    }
    __syncthreads();
#endif

    const int n0  = ct * CHUNK_COLS;
    const int col = n0 + nlane;
    float cn = colnorm[col];
    const char* lp = bb + nlane * COLB_LDS + chalf * 2;

    v8f acc = {};
#pragma unroll
    for (int kt = 0; kt < 8; ++kt) {
      v8bf h0 = *(const v8bf*)(lp + kt * 64);
      v8bf h1 = *(const v8bf*)(lp + kt * 64 + 32);
      v8bf l0 = *(const v8bf*)(lp + 512 + kt * 64);
      v8bf l1 = *(const v8bf*)(lp + 512 + kt * 64 + 32);
      v16bf bh = __builtin_shufflevector(h0, h1, 0, 1, 2, 3, 4, 5, 6, 7,
                                         8, 9, 10, 11, 12, 13, 14, 15);
      v16bf bl = __builtin_shufflevector(l0, l1, 0, 1, 2, 3, 4, 5, 6, 7,
                                         8, 9, 10, 11, 12, 13, 14, 15);
      acc = __builtin_amdgcn_wmma_f32_16x16x32_bf16(false, a_hi[kt], false, bh,
                                                    (short)0, acc, false, false);
      acc = __builtin_amdgcn_wmma_f32_16x16x32_bf16(false, a_lo[kt], false, bh,
                                                    (short)0, acc, false, false);
      acc = __builtin_amdgcn_wmma_f32_16x16x32_bf16(false, a_hi[kt], false, bl,
                                                    (short)0, acc, false, false);
    }
#if VQ_SCHED
    // Pin pipeline: DS-read groups lead WMMA groups by one kt stage.
    // DS4(kt0) DS4(kt1) [WMMA3(kt) DS4(kt+2)]x6 WMMA3(kt6) WMMA3(kt7)
    __builtin_amdgcn_sched_group_barrier(0x100, 4, 0);   // DS reads kt0
    __builtin_amdgcn_sched_group_barrier(0x100, 4, 0);   // DS reads kt1
#pragma unroll
    for (int g = 0; g < 6; ++g) {
      __builtin_amdgcn_sched_group_barrier(0x008, 3, 0); // WMMA kt g
      __builtin_amdgcn_sched_group_barrier(0x100, 4, 0); // DS reads kt g+2
    }
    __builtin_amdgcn_sched_group_barrier(0x008, 3, 0);   // WMMA kt6
    __builtin_amdgcn_sched_group_barrier(0x008, 3, 0);   // WMMA kt7
#endif
#pragma unroll
    for (int v = 0; v < 8; ++v) {
      float d = cn - 2.0f * acc[v];     // argmin key (||f||^2 is row-constant)
      if (d < minv[v]) { minv[v] = d; mini[v] = col; }
    }
#if !VQ_TDM
    __syncthreads();                    // protect buffer before next overwrite
#endif
  }

  // ---- argmin reduce across the 16 lanes of each half (rows 0-7 / 8-15)
#pragma unroll
  for (int off = 8; off >= 1; off >>= 1) {
#pragma unroll
    for (int v = 0; v < 8; ++v) {
      float ov = __shfl_xor(minv[v], off, 32);
      int   oi = __shfl_xor(mini[v], off, 32);
      if (ov < minv[v] || (ov == minv[v] && oi < mini[v])) { minv[v] = ov; mini[v] = oi; }
    }
  }
  if ((lane & 15) == 0) {
    int base = wave * 16 + (lane >> 4) * 8;
#pragma unroll
    for (int v = 0; v < 8; ++v) resIdx[base + v] = mini[v];
  }
  __syncthreads();

  // ---- outputs: indices + histogram
  if (t < RPB) {
    int idx = resIdx[t];
    out[IND_OFF + R0 + t] = (float)idx;
    atomicAdd(&counts[idx], 1.0f);
  }

  // ---- gather codebook -> quantize [B,C,H,W], accumulate SSE
  float sse = 0.0f;
  for (int i = 0; i < (RPB * DIM) / 256; ++i) {
    int linear = i * 256 + t;
    int r = linear & (RPB - 1);
    int c = linear >> 7;
    int idx = resIdx[r];
    float q = e[(size_t)c * KCODES + idx];
    size_t gaddr = ((size_t)(b * DIM + c)) * HW + hw0 + r;
    float xv = x[gaddr];
    out[QUANT_OFF + gaddr] = q;
    float d = q - xv;
    sse += d * d;
  }
  red[t] = sse;
  __syncthreads();
  for (int s = 128; s > 0; s >>= 1) {
    if (t < s) red[t] += red[t + s];
    __syncthreads();
  }
  if (t == 0) atomicAdd(sse_acc, red[0]);
}

// ---------------------------------------------------------------- finalize
__global__ void vq_finalize(const float* __restrict__ counts,
                            const float* __restrict__ sse,
                            float* __restrict__ out) {
  __shared__ float red[256];
  int t = threadIdx.x;
  float s = 0.0f;
  for (int k = t; k < KCODES; k += 256) {
    float p = counts[k] * (1.0f / (float)NROWS);
    s += p * __logf(p + 1e-10f);
  }
  red[t] = s;
  __syncthreads();
  for (int st = 128; st > 0; st >>= 1) {
    if (t < st) red[t] += red[t + st];
    __syncthreads();
  }
  if (t == 0) {
    out[PERP_OFF] = __expf(-red[0]);
    out[DIFF_OFF] = sse[0] * (1.0f / 4194304.0f);
  }
}

// ---------------------------------------------------------------- launch
extern "C" void kernel_launch(void* const* d_in, const int* in_sizes, int n_in,
                              void* d_out, int out_size, void* d_ws, size_t ws_size,
                              hipStream_t stream) {
  const float* x = (const float*)d_in[0];   // (16,256,32,32) f32
  const float* e = (const float*)d_in[1];   // (1,1,256,8192) f32
  float* out = (float*)d_out;

  char* ws = (char*)d_ws;
  __bf16* etc    = (__bf16*)(ws);                            // 8 MB combined hi|lo
  float*  cnorm  = (float*)(ws + (size_t)KCODES * COLB);     // 32 KB
  float*  counts = cnorm + KCODES;                           // 32 KB
  float*  sse    = counts + KCODES;                          // 4 B

  vq_init<<<(KCODES + 255) / 256, 256, 0, stream>>>(counts, sse);
  dim3 tg(KCODES / 32, DIM / 32);
  vq_prep_codes<<<tg, 256, 0, stream>>>(e, etc);
  vq_colnorm<<<KCODES / 256, 256, 0, stream>>>(e, cnorm);
  const size_t dynsmem = 2 * CHUNK_LDS + 16640 + 1024 + 512; // 51456 B
  vq_main<<<NROWS / RPB, 256, dynsmem, stream>>>(x, e, etc, cnorm,
                                                 out, counts, sse);
  vq_finalize<<<1, 256, 0, stream>>>(counts, sse, out);
}
